// FSAttention_11476152615109
// MI455X (gfx1250) — matrix operand order repaired and checked by reading, not compiled, not run on hardware
//
#include <hip/hip_runtime.h>
#include <hip/hip_bf16.h>

// ---------------------------------------------------------------------------
// Problem constants (from reference)
// ---------------------------------------------------------------------------
#define DIMX   1024
#define NSEQ   2048
#define BATCH  2
#define HEADS  16
#define DHEAD  64
#define NROWS  (BATCH * NSEQ)          // 4096
#define ATT_SCALE 0.125f               // 64^-0.5

typedef __bf16 bf16;
typedef __attribute__((ext_vector_type(16))) __bf16 v16bf;
typedef __attribute__((ext_vector_type(8)))  __bf16 v8bf;
typedef __attribute__((ext_vector_type(8)))  float  v8f;

static __device__ __forceinline__ v8f vzero8() {
    v8f r;
#pragma unroll
    for (int i = 0; i < 8; ++i) r[i] = 0.0f;
    return r;
}

static __device__ __forceinline__ v16bf pack16(v8bf lo, v8bf hi) {
    v16bf r;
#pragma unroll
    for (int i = 0; i < 8; ++i) { r[i] = lo[i]; r[i + 8] = hi[i]; }
    return r;
}

static __device__ __forceinline__ v8f wmma_bf16(v16bf a, v16bf b, v8f c) {
    return __builtin_amdgcn_wmma_f32_16x16x32_bf16(
        /*neg_a=*/false, a, /*neg_b=*/false, b,
        /*c_mod=*/(short)0, c, /*reuse_a=*/false, /*reuse_b=*/false);
}

// ---------------------------------------------------------------------------
// Kernel 1: fused QKV projection  C = X @ W^T + b  (fp32 in, bf16 out to ws)
//   block 256 threads = 8 waves; block tile 128x64; wave tile 32x32;
//   K staged 64 deep (two 32-wide WMMA sub-steps per barrier).
//   gridDim.z selects (Wq,bq,Q*scale) / (Wk,bk,K) / (Wv,bv,V)
// ---------------------------------------------------------------------------
#define BLK_M 128
#define BLK_N 64
#define BLK_K 64

__global__ __launch_bounds__(256)
void qkv_gemm_kernel(const float* __restrict__ X,
                     const float* __restrict__ Wq, const float* __restrict__ bq,
                     const float* __restrict__ Wk, const float* __restrict__ bk,
                     const float* __restrict__ Wv, const float* __restrict__ bv,
                     bf16* __restrict__ ws) {
    const int z = blockIdx.z;
    const float* W    = (z == 0) ? Wq : (z == 1) ? Wk : Wv;
    const float* bias = (z == 0) ? bq : (z == 1) ? bk : bv;
    bf16* out = ws + (size_t)z * (size_t)NROWS * (size_t)DIMX;
    const float scale = (z == 0) ? ATT_SCALE : 1.0f;

    __shared__ bf16 lX[BLK_M][BLK_K + 8];        // 128 x 72  (row-major m,k) 18.0KB
    __shared__ bf16 lWT[BLK_K][BLK_N + 8];       //  64 x 72  (transposed: k,n) 9.0KB

    const int t    = threadIdx.x;
    const int wave = t >> 5;
    const int lane = t & 31;
    const int wm   = wave >> 1;                  // 0..3
    const int wn   = wave & 1;                   // 0..1
    const int m    = lane & 15;
    const int g    = lane >> 4;

    const int m0 = blockIdx.x * BLK_M;
    const int n0 = blockIdx.y * BLK_N;

    v8f acc[2][2];
#pragma unroll
    for (int i = 0; i < 2; ++i)
#pragma unroll
        for (int j = 0; j < 2; ++j) acc[i][j] = vzero8();

    // staging roles
    const int sx_row  = t >> 1;                  // 0..127
    const int sx_half = t & 1;                   // 0..1  (32 floats each)
    const int sw_n    = t >> 2;                  // 0..63
    const int sw_kc   = t & 3;                   // 0..3  (16 floats each)

    for (int k0 = 0; k0 < DIMX; k0 += BLK_K) {
        if (k0) __syncthreads();
        // ---- stage X tile (convert fp32 -> bf16) ----
        {
            const float4* p = (const float4*)(X + (size_t)(m0 + sx_row) * DIMX + k0 + sx_half * 32);
            float tf[32];
#pragma unroll
            for (int q = 0; q < 8; ++q) {
                float4 f = p[q];
                tf[q * 4 + 0] = f.x; tf[q * 4 + 1] = f.y;
                tf[q * 4 + 2] = f.z; tf[q * 4 + 3] = f.w;
            }
#pragma unroll
            for (int q = 0; q < 4; ++q) {
                v8bf v;
#pragma unroll
                for (int i = 0; i < 8; ++i) v[i] = (bf16)tf[q * 8 + i];
                *(v8bf*)&lX[sx_row][sx_half * 32 + q * 8] = v;
            }
            if (k0 + BLK_K < DIMX)
                __builtin_prefetch(X + (size_t)(m0 + sx_row) * DIMX + k0 + BLK_K + sx_half * 32, 0, 1);
        }
        // ---- stage W tile transposed: lWT[k][n] = W[n0+n][k0+k] ----
        {
            const float4* p = (const float4*)(W + (size_t)(n0 + sw_n) * DIMX + k0 + sw_kc * 16);
            float tf[16];
#pragma unroll
            for (int q = 0; q < 4; ++q) {
                float4 f = p[q];
                tf[q * 4 + 0] = f.x; tf[q * 4 + 1] = f.y;
                tf[q * 4 + 2] = f.z; tf[q * 4 + 3] = f.w;
            }
#pragma unroll
            for (int kk = 0; kk < 16; ++kk)
                lWT[sw_kc * 16 + kk][sw_n] = (bf16)tf[kk];
        }
        __syncthreads();

        // ---- fragments + WMMA (two 32-wide K sub-steps) ----
#pragma unroll
        for (int ks = 0; ks < 2; ++ks) {
            v16bf afrag[2], bfrag[2];
#pragma unroll
            for (int ms = 0; ms < 2; ++ms) {
                const int row = wm * 32 + ms * 16 + m;
                v8bf lo = *(const v8bf*)&lX[row][ks * 32 + g * 8];
                v8bf hi = *(const v8bf*)&lX[row][ks * 32 + 16 + g * 8];
                afrag[ms] = pack16(lo, hi);
            }
#pragma unroll
            for (int ns = 0; ns < 2; ++ns) {
                const int col = wn * 32 + ns * 16;
                v16bf bb;
#pragma unroll
                for (int i = 0; i < 8; ++i) {
                    bb[i]     = lWT[ks * 32 + g * 8 + i][col + m];
                    bb[i + 8] = lWT[ks * 32 + 16 + g * 8 + i][col + m];
                }
                bfrag[ns] = bb;
            }
#pragma unroll
            for (int ms = 0; ms < 2; ++ms)
#pragma unroll
                for (int ns = 0; ns < 2; ++ns)
                    acc[ms][ns] = wmma_bf16(afrag[ms], bfrag[ns], acc[ms][ns]);
        }
    }

    // ---- epilogue: + bias, * scale, store bf16 ----
#pragma unroll
    for (int ms = 0; ms < 2; ++ms) {
#pragma unroll
        for (int ns = 0; ns < 2; ++ns) {
            const int col = n0 + wn * 32 + ns * 16 + (lane & 15);
            const float bv = bias[col];
#pragma unroll
            for (int jv = 0; jv < 8; ++jv) {
                const int row = m0 + wm * 32 + ms * 16 + jv + 8 * g;
                out[(size_t)row * DIMX + col] = (bf16)((acc[ms][ns][jv] + bv) * scale);
            }
        }
    }
}

// ---------------------------------------------------------------------------
// Kernel 2: flash-style attention per (64 q-rows, head, batch)
//   block 128 threads = 4 waves; each wave owns 16 q rows.
//   j-loop over keys in steps of 64; K staged transposed (manual, needs
//   transpose), V staged with GLOBAL_LOAD_ASYNC_TO_LDS_B128 (pure copy).
// ---------------------------------------------------------------------------
#define JT 64

__global__ __launch_bounds__(128)
void attention_kernel(const bf16* __restrict__ ws, float* __restrict__ out) {
    const bf16* Q = ws;
    const bf16* K = ws + (size_t)NROWS * DIMX;
    const bf16* V = ws + 2 * (size_t)NROWS * DIMX;

    __shared__ bf16 lKT[DHEAD][JT + 8];          // 64 x 72  (d, j)   9.0KB
    __shared__ bf16 lV[JT][DHEAD + 8];           // 64 x 72  (j, d)   9.0KB
    __shared__ bf16 lP[4][16][JT + 8];           // wave, m, j        9.0KB

    const int t    = threadIdx.x;
    const int wave = t >> 5;                     // 0..3
    const int lane = t & 31;
    const int m    = lane & 15;
    const int g    = lane >> 4;

    const int q0 = blockIdx.x * 64;
    const int h  = blockIdx.y;
    const int b  = blockIdx.z;
    const int qrowbase = b * NSEQ + q0 + wave * 16;

    // ---- load Q A-fragments once (bf16, already scaled) ----
    v16bf qa[2];
#pragma unroll
    for (int dh = 0; dh < 2; ++dh) {
        const size_t base = (size_t)(qrowbase + m) * DIMX + h * DHEAD + dh * 32 + g * 8;
        v8bf lo = *(const v8bf*)(Q + base);
        v8bf hi = *(const v8bf*)(Q + base + 16);
        qa[dh] = pack16(lo, hi);
    }

    v8f acc[4];
#pragma unroll
    for (int i = 0; i < 4; ++i) acc[i] = vzero8();
    float rmax[8], rsum[8];
#pragma unroll
    for (int jv = 0; jv < 8; ++jv) { rmax[jv] = -1e30f; rsum[jv] = 0.0f; }

    const int skey  = t >> 1;                    // 0..63
    const int shalf = t & 1;                     // 0..1  (32 elems each)

    for (int j0 = 0; j0 < NSEQ; j0 += JT) {
        __syncthreads();
        // ---- stage K (manual transpose) + V (async copy, no transpose) ----
        {
            const size_t base = (size_t)(b * NSEQ + j0 + skey) * DIMX + h * DHEAD + shalf * 32;
            // V: global -> LDS direct, tracked by ASYNCcnt (64B per thread)
            unsigned lva = (unsigned)(size_t)&lV[skey][shalf * 32];
            const bf16* gva = V + base;
            asm volatile(
                "global_load_async_to_lds_b128 %0, %1, off\n\t"
                "global_load_async_to_lds_b128 %0, %1, off offset:16\n\t"
                "global_load_async_to_lds_b128 %0, %1, off offset:32\n\t"
                "global_load_async_to_lds_b128 %0, %1, off offset:48"
                :: "v"(lva), "v"(gva) : "memory");
            // K: load 32 bf16, scatter-transpose into lKT[d][j]
            v8bf k0v = *(const v8bf*)(K + base);
            v8bf k1v = *(const v8bf*)(K + base + 8);
            v8bf k2v = *(const v8bf*)(K + base + 16);
            v8bf k3v = *(const v8bf*)(K + base + 24);
#pragma unroll
            for (int i = 0; i < 8; ++i) {
                lKT[shalf * 32 + i][skey]      = k0v[i];
                lKT[shalf * 32 + 8 + i][skey]  = k1v[i];
                lKT[shalf * 32 + 16 + i][skey] = k2v[i];
                lKT[shalf * 32 + 24 + i][skey] = k3v[i];
            }
        }
        asm volatile("s_wait_asynccnt 0" ::: "memory");
        __syncthreads();

        // ---- S = Q @ K^T  (16 x 64 scores per wave) ----
        v8f s[4];
#pragma unroll
        for (int c = 0; c < 4; ++c) s[c] = vzero8();
#pragma unroll
        for (int dh = 0; dh < 2; ++dh) {
            const int krow = dh * 32 + lane;
#pragma unroll
            for (int c = 0; c < 4; ++c) {
                v16bf kb;
#pragma unroll
                for (int i = 0; i < 8; ++i) {
                    kb[i]     = lKT[dh * 32 + g * 8 + i][c * 16 + m];
                    kb[i + 8] = lKT[dh * 32 + 16 + g * 8 + i][c * 16 + m];
                }
                s[c] = wmma_bf16(qa[dh], kb, s[c]);
            }
        }

        // ---- online softmax (rows private to this wave); exp in place ----
        float corr[8];
#pragma unroll
        for (int jv = 0; jv < 8; ++jv) {
            float c01 = fmaxf(s[0][jv], s[1][jv]);
            float c23 = fmaxf(s[2][jv], s[3][jv]);
            float cm = fmaxf(c01, c23);
            cm = fmaxf(cm, __shfl_xor(cm, 1));
            cm = fmaxf(cm, __shfl_xor(cm, 2));
            cm = fmaxf(cm, __shfl_xor(cm, 4));
            cm = fmaxf(cm, __shfl_xor(cm, 8));
            const float nm = fmaxf(rmax[jv], cm);
            corr[jv] = __expf(rmax[jv] - nm);
            rmax[jv] = nm;
            float ps = 0.0f;
#pragma unroll
            for (int c = 0; c < 4; ++c) {
                const float p = __expf(s[c][jv] - nm);
                s[c][jv] = p;
                ps += p;
            }
            ps += __shfl_xor(ps, 1);
            ps += __shfl_xor(ps, 2);
            ps += __shfl_xor(ps, 4);
            ps += __shfl_xor(ps, 8);
            rsum[jv] = rsum[jv] * corr[jv] + ps;
        }
#pragma unroll
        for (int t4 = 0; t4 < 4; ++t4)
#pragma unroll
            for (int jv = 0; jv < 8; ++jv) acc[t4][jv] *= corr[jv];

        // ---- P (C-layout) -> LDS -> A-layout fragments ----
#pragma unroll
        for (int c = 0; c < 4; ++c)
#pragma unroll
            for (int jv = 0; jv < 8; ++jv)
                lP[wave][jv + 8 * g][c * 16 + (lane & 15)] = (bf16)s[c][jv];
        asm volatile("s_wait_dscnt 0" ::: "memory");

        // ---- out += P @ V  (contraction over 64 keys = two A frags) ----
#pragma unroll
        for (int kh = 0; kh < 2; ++kh) {
            v8bf plo = *(const v8bf*)&lP[wave][m][kh * 32 + g * 8];
            v8bf phi = *(const v8bf*)&lP[wave][m][kh * 32 + 16 + g * 8];
            v16bf pa = pack16(plo, phi);
#pragma unroll
            for (int dsub = 0; dsub < 4; ++dsub) {
                v16bf vb;
#pragma unroll
                for (int i = 0; i < 8; ++i) {
                    vb[i]     = lV[kh * 32 + g * 8 + i][dsub * 16 + m];
                    vb[i + 8] = lV[kh * 32 + 16 + g * 8 + i][dsub * 16 + m];
                }
                acc[dsub] = wmma_bf16(pa, vb, acc[dsub]);
            }
        }
    }

    // ---- normalize and store fp32 output ----
#pragma unroll
    for (int jv = 0; jv < 8; ++jv) {
        const float inv = 1.0f / rsum[jv];
        const int row = b * NSEQ + q0 + wave * 16 + jv + 8 * g;
#pragma unroll
        for (int dsub = 0; dsub < 4; ++dsub) {
            out[(size_t)row * DIMX + h * DHEAD + dsub * 16 + (lane & 15)] =
                acc[dsub][jv] * inv;
        }
    }
}

// ---------------------------------------------------------------------------
// Launcher
// ---------------------------------------------------------------------------
extern "C" void kernel_launch(void* const* d_in, const int* in_sizes, int n_in,
                              void* d_out, int out_size, void* d_ws, size_t ws_size,
                              hipStream_t stream) {
    (void)in_sizes; (void)n_in; (void)out_size; (void)ws_size;
    const float* x  = (const float*)d_in[0];
    const float* Wq = (const float*)d_in[1];
    const float* bq = (const float*)d_in[2];
    const float* Wk = (const float*)d_in[3];
    const float* bk = (const float*)d_in[4];
    const float* Wv = (const float*)d_in[5];
    const float* bv = (const float*)d_in[6];
    bf16* ws = (bf16*)d_ws;                 // Q | K | V, each 4096*1024 bf16
    float* out = (float*)d_out;

    dim3 gGemm(NROWS / BLK_M, DIMX / BLK_N, 3);     // 32 x 16 x 3
    hipLaunchKernelGGL(qkv_gemm_kernel, gGemm, dim3(256), 0, stream,
                       x, Wq, bq, Wk, bk, Wv, bv, ws);

    dim3 gAtt(NSEQ / 64, HEADS, BATCH);             // 32 x 16 x 2
    hipLaunchKernelGGL(attention_kernel, gAtt, dim3(128), 0, stream, ws, out);
}
